// GroupedMultiQueryAttention_66279935312391
// MI455X (gfx1250) — matrix operand order repaired and checked by reading, not compiled, not run on hardware
//
#include <hip/hip_runtime.h>
#include <hip/hip_bf16.h>

// ---------------- problem constants ----------------
#define B_      2
#define S_      2048
#define DM      2048
#define NH      16
#define NKV     4
#define DEPTH_  128
#define MAXB    2
#define MAXS    4096

typedef __attribute__((ext_vector_type(16))) __bf16 v16bf;
typedef __attribute__((ext_vector_type(8)))  float  v8f;

static __device__ __forceinline__ v8f wmma_bf(v16bf a, v16bf b, v8f c) {
  // D = A(16x32 bf16) x B(32x16 bf16) + C(16x16 f32)
  return __builtin_amdgcn_wmma_f32_16x16x32_bf16(false, a, false, b, (short)0, c,
                                                 false, false);
}

static __device__ __forceinline__ void split2(float x, __bf16& h, __bf16& l) {
  h = (__bf16)x;
  l = (__bf16)(x - (float)h);
}

// ---------------- per-head column sums of a weight matrix ----------------
// W: [K, H*128] row-major.  out[c*H + h] = sum_{d<128} W[c, h*128+d]
__global__ void colsum_kernel(const float* __restrict__ W, float* __restrict__ out,
                              int K, int H) {
  int i = blockIdx.x * blockDim.x + threadIdx.x;
  if (i >= K * H) return;
  int c = i / H, h = i % H;
  const float* p = W + (size_t)c * (H * DEPTH_) + h * DEPTH_;
  float s = 0.f;
#pragma unroll 8
  for (int d = 0; d < DEPTH_; ++d) s += p[d];
  out[i] = s;
}

// Sq[(b*S+s)*NH + h] = q_in[row,:] . wqsum[:,h]
__global__ void rowsum_q_kernel(const float* __restrict__ X,
                                const float* __restrict__ ws,
                                float* __restrict__ Sq) {
  int i = blockIdx.x * blockDim.x + threadIdx.x;
  if (i >= B_ * S_ * NH) return;
  int r = i / NH, h = i % NH;
  const float* x = X + (size_t)r * DM;
  float s = 0.f;
#pragma unroll 4
  for (int c = 0; c < DM; ++c) s += x[c] * ws[c * NH + h];
  Sq[i] = s;
}

// Skc[(b*NKV+j)*S_ + s] = k_in[row,:] . wksum[:,j]   (transposed, contiguous in s)
__global__ void rowsum_k_kernel(const float* __restrict__ X,
                                const float* __restrict__ ws,
                                float* __restrict__ Skc) {
  int i = blockIdx.x * blockDim.x + threadIdx.x;
  if (i >= B_ * S_ * NKV) return;
  int r = i / NKV, j = i % NKV;
  int b = r / S_, sidx = r % S_;
  const float* x = X + (size_t)r * DM;
  float s = 0.f;
#pragma unroll 4
  for (int c = 0; c < DM; ++c) s += x[c] * ws[c * NKV + j];
  Skc[(b * NKV + j) * S_ + sidx] = s;
}

// per (b,kv) min/max of Sk over the sequence (for exact softmax max-subtraction)
__global__ void minmax_kernel(const float* __restrict__ Skc,
                              float* __restrict__ mx, float* __restrict__ mn) {
  __shared__ float smax[256], smin[256];
  int g = blockIdx.x;  // 0..B_*NKV-1
  const float* p = Skc + (size_t)g * S_;
  float a = -3.4e38f, b = 3.4e38f;
  for (int s = threadIdx.x; s < S_; s += 256) {
    float v = p[s];
    a = fmaxf(a, v);
    b = fminf(b, v);
  }
  smax[threadIdx.x] = a; smin[threadIdx.x] = b;
  __syncthreads();
  for (int o = 128; o > 0; o >>= 1) {
    if ((int)threadIdx.x < o) {
      smax[threadIdx.x] = fmaxf(smax[threadIdx.x], smax[threadIdx.x + o]);
      smin[threadIdx.x] = fminf(smin[threadIdx.x], smin[threadIdx.x + o]);
    }
    __syncthreads();
  }
  if (threadIdx.x == 0) { mx[g] = smax[0]; mn[g] = smin[0]; }
}

// W [K,N] f32 -> Bt_hi/Bt_lo [N,K] bf16 (transposed + hi/lo split)
__global__ void convertBt_kernel(const float* __restrict__ W,
                                 __bf16* __restrict__ hi, __bf16* __restrict__ lo,
                                 int K, int N) {
  int i = blockIdx.x * blockDim.x + threadIdx.x;
  if (i >= K * N) return;
  int n = i / K, k = i % K;
  float v = W[(size_t)k * N + n];
  __bf16 h, l; split2(v, h, l);
  hi[i] = h; lo[i] = l;
}

// v_cache region [MAXB,MAXS,NKV,128] -> Vt_hi/lo [B,NKV,128,S] bf16
__global__ void convert_vt_kernel(const float* __restrict__ vc,
                                  __bf16* __restrict__ hi, __bf16* __restrict__ lo) {
  int i = blockIdx.x * blockDim.x + threadIdx.x;
  if (i >= B_ * NKV * DEPTH_ * S_) return;
  int s = i % S_;
  int d = (i / S_) % DEPTH_;
  int kv = (i / (S_ * DEPTH_)) % NKV;
  int b = i / (S_ * DEPTH_ * NKV);
  float v = vc[(((size_t)b * MAXS + s) * NKV + kv) * DEPTH_ + d];
  __bf16 h, l; split2(v, h, l);
  hi[i] = h; lo[i] = l;
}

// k_cache broadcast-fill + zero-pad of both caches (s >= S_)
__global__ void fill_cache_kernel(const float* __restrict__ Skc,
                                  float* __restrict__ kc, float* __restrict__ vc) {
  int i = blockIdx.x * blockDim.x + threadIdx.x;
  const int TOT = MAXB * MAXS * NKV * DEPTH_;
  if (i >= TOT) return;
  int b = i / (MAXS * NKV * DEPTH_);
  int rem = i % (MAXS * NKV * DEPTH_);
  int s = rem / (NKV * DEPTH_);
  int kv = (rem / DEPTH_) % NKV;
  if (s < S_) {
    kc[i] = Skc[(b * NKV + kv) * S_ + s];   // rotate(k) broadcasts the row-sum
  } else {
    kc[i] = 0.f;
    vc[i] = 0.f;
  }
}

// ---------------- generic bf16x3 WMMA GEMM:  C = A @ B ----------------
// A [M,K] f32 row-major (split to bf16 hi/lo on the fly)
// Bth/Btl [N,K] bf16 (pre-transposed, hi/lo)
// B fragments are double-buffered across k-steps so global_load clauses for
// step i+1 are in flight during step i's 12 WMMAs (K % 64 == 0 always here).
// remap==1: rows scattered into the KV-cache layout (row b*MAXS + s), ld = N
__global__ __launch_bounds__(256)
void gemm_bf16x3_kernel(const float* __restrict__ A,
                        const __bf16* __restrict__ Bth,
                        const __bf16* __restrict__ Btl,
                        float* __restrict__ C, int M, int N, int K, int remap) {
  const int lane = threadIdx.x & 31;
  const int wave = threadIdx.x >> 5;
  const int hlf  = lane >> 4;      // wave32 half: selects K sub-block
  const int ln   = lane & 15;
  const int m0 = blockIdx.y * 128 + wave * 16;
  const int n0 = blockIdx.x * 64;

  v8f acc[4] = {};
  const float* arow = A + (size_t)(m0 + ln) * K;

  auto loadB4 = [&](v16bf (&bh)[4], v16bf (&bl)[4], int kk) {
#pragma unroll
    for (int t = 0; t < 4; ++t) {
      size_t bo = (size_t)(n0 + t * 16 + ln) * K + kk + 16 * hlf;
      bh[t] = *(const v16bf*)(Bth + bo);
      bl[t] = *(const v16bf*)(Btl + bo);
    }
  };
  auto loadSplitA = [&](v16bf& ah, v16bf& al, int kk) {
    float4 f0 = *(const float4*)(arow + kk + hlf * 16);
    float4 f1 = *(const float4*)(arow + kk + hlf * 16 + 4);
    float4 f2 = *(const float4*)(arow + kk + hlf * 16 + 8);
    float4 f3 = *(const float4*)(arow + kk + hlf * 16 + 12);
    float af[16] = {f0.x, f0.y, f0.z, f0.w, f1.x, f1.y, f1.z, f1.w,
                    f2.x, f2.y, f2.z, f2.w, f3.x, f3.y, f3.z, f3.w};
#pragma unroll
    for (int i = 0; i < 16; ++i) {
      __bf16 h, l; split2(af[i], h, l);
      ah[i] = h; al[i] = l;
    }
  };

  v16bf bh0[4], bl0[4], bh1[4], bl1[4], ah0, al0, ah1, al1;
  loadB4(bh0, bl0, 0);
  for (int k0 = 0; k0 < K; k0 += 64) {
    __builtin_prefetch(arow + k0 + 256, 0, 1);
    loadB4(bh1, bl1, k0 + 32);             // in flight during first WMMA block
    loadSplitA(ah0, al0, k0);
#pragma unroll
    for (int t = 0; t < 4; ++t) {
      acc[t] = wmma_bf(ah0, bh0[t], acc[t]);
      acc[t] = wmma_bf(al0, bh0[t], acc[t]);
      acc[t] = wmma_bf(ah0, bl0[t], acc[t]);
    }
    int kn = (k0 + 64 < K) ? k0 + 64 : 0;  // dead prefetch on last iter, branch-free
    loadB4(bh0, bl0, kn);                  // in flight during second WMMA block
    loadSplitA(ah1, al1, k0 + 32);
#pragma unroll
    for (int t = 0; t < 4; ++t) {
      acc[t] = wmma_bf(ah1, bh1[t], acc[t]);
      acc[t] = wmma_bf(al1, bh1[t], acc[t]);
      acc[t] = wmma_bf(ah1, bl1[t], acc[t]);
    }
  }

#pragma unroll
  for (int t = 0; t < 4; ++t) {
#pragma unroll
    for (int r = 0; r < 8; ++r) {
      int row = m0 + r + 8 * hlf;           // C layout: VGPR r -> row r + 8*half
      int col = n0 + t * 16 + ln;
      size_t off;
      if (remap) {
        int b = row >> 11, s = row & (S_ - 1);
        off = ((size_t)b * MAXS + s) * N + col;
      } else {
        off = (size_t)row * N + col;
      }
      C[off] = acc[t][r];
    }
  }
}

// ---------------- fused rank-1 softmax + attn@V (WMMA) ----------------
// per wave: 16 query rows x full DEPTH=128 for one (b,h) head.
// All 16 V-fragment loads of a k-step issue as one clause up front; the
// exp/split transcendental work overlaps their latency before the 24 WMMAs.
__global__ __launch_bounds__(256)
void attention_kernel(const float* __restrict__ Sq, const float* __restrict__ Skc,
                      const float* __restrict__ skmax, const float* __restrict__ skmin,
                      const __bf16* __restrict__ Vth, const __bf16* __restrict__ Vtl,
                      float* __restrict__ Out) {
  const int wid = blockIdx.x * 8 + (threadIdx.x >> 5);  // 0..4095
  const int lane = threadIdx.x & 31;
  const int QT = S_ / 16;
  int b = wid / (NH * QT);
  int rem = wid % (NH * QT);
  int h = rem / QT;
  int qt = rem % QT;
  int kv = h / (NH / NKV);
  const int hlf = lane >> 4, ln = lane & 15;

  // this lane generates softmax weights for query row m = ln
  float aq = 128.f * Sq[(size_t)(b * S_ + qt * 16 + ln) * NH + h];  // DEPTH * Sq
  float mx = skmax[b * NKV + kv], mn = skmin[b * NKV + kv];
  float mq = (aq >= 0.f) ? aq * mx : aq * mn;  // exact max_k(aq*Sk)
  const float* skp = Skc + (size_t)(b * NKV + kv) * S_;
  const __bf16* vbh = Vth + (size_t)((b * NKV + kv) * DEPTH_) * S_;
  const __bf16* vbl = Vtl + (size_t)((b * NKV + kv) * DEPTH_) * S_;

  v8f acc[8] = {};
  float z = 0.f;

  for (int k0 = 0; k0 < S_; k0 += 32) {
    // ---- issue everything memory-bound first (one clause) ----
    float4 s0 = *(const float4*)(skp + k0 + hlf * 16);
    float4 s1 = *(const float4*)(skp + k0 + hlf * 16 + 4);
    float4 s2 = *(const float4*)(skp + k0 + hlf * 16 + 8);
    float4 s3 = *(const float4*)(skp + k0 + hlf * 16 + 12);
    v16bf bh[8], bl[8];
#pragma unroll
    for (int t = 0; t < 8; ++t) {
      size_t bo = (size_t)(t * 16 + ln) * S_ + k0 + 16 * hlf;
      bh[t] = *(const v16bf*)(vbh + bo);
      bl[t] = *(const v16bf*)(vbl + bo);
    }
    // ---- transcendental / split work overlaps the loads ----
    float sk[16] = {s0.x, s0.y, s0.z, s0.w, s1.x, s1.y, s1.z, s1.w,
                    s2.x, s2.y, s2.z, s2.w, s3.x, s3.y, s3.z, s3.w};
    v16bf ah, al;
#pragma unroll
    for (int i = 0; i < 16; ++i) {
      float p = __expf(fmaf(aq, sk[i], -mq));  // <= 1, underflows to 0 far away
      z += p;                                  // partial denominator for row ln
      __bf16 ph, pl; split2(p, ph, pl);
      ah[i] = ph; al[i] = pl;
    }
    // ---- matrix pipe ----
#pragma unroll
    for (int t = 0; t < 8; ++t) {
      acc[t] = wmma_bf(ah, bh[t], acc[t]);
      acc[t] = wmma_bf(al, bh[t], acc[t]);
      acc[t] = wmma_bf(ah, bl[t], acc[t]);
    }
  }

  // lanes L and L+16 covered complementary K halves of row L&15
  z += __shfl_xor(z, 16, 32);   // now lane L holds full Z for row (L & 15)

#pragma unroll
  for (int r = 0; r < 8; ++r) {
    float zr = __shfl(z, r + 8 * hlf, 32);   // Z of output row r + 8*half
    float inv = 1.f / zr;
    int q = qt * 16 + r + 8 * hlf;
    size_t rowoff = (size_t)(b * S_ + q) * DM + (size_t)h * DEPTH_;
#pragma unroll
    for (int t = 0; t < 8; ++t) Out[rowoff + t * 16 + ln] = acc[t][r] * inv;
  }
}

// ---------------- launcher ----------------
extern "C" void kernel_launch(void* const* d_in, const int* in_sizes, int n_in,
                              void* d_out, int out_size, void* d_ws, size_t ws_size,
                              hipStream_t stream) {
  (void)in_sizes; (void)n_in; (void)out_size; (void)ws_size;
  const float* q_in = (const float*)d_in[0];
  const float* k_in = (const float*)d_in[1];
  const float* v_in = (const float*)d_in[2];
  const float* Wq = (const float*)d_in[4];
  const float* Wk = (const float*)d_in[5];
  const float* Wv = (const float*)d_in[6];
  const float* Wo = (const float*)d_in[7];

  float* out    = (float*)d_out;                 // [B,S,DM]
  float* kcache = out + (size_t)B_ * S_ * DM;    // [MAXB,MAXS,NKV,128]
  float* vcache = kcache + (size_t)MAXB * MAXS * NKV * DEPTH_;

  // workspace carving (256B aligned segments)
  char* w = (char*)d_ws;
  size_t off = 0;
  auto take = [&](size_t bytes) { void* p = w + off; off = (off + bytes + 255) & ~(size_t)255; return p; };
  float*  wqsum = (float*)take(DM * NH * 4);
  float*  wksum = (float*)take(DM * NKV * 4);
  float*  Sq    = (float*)take((size_t)B_ * S_ * NH * 4);
  float*  Skc   = (float*)take((size_t)B_ * NKV * S_ * 4);
  float*  skmax = (float*)take(B_ * NKV * 4);
  float*  skmin = (float*)take(B_ * NKV * 4);
  __bf16* wvt_h = (__bf16*)take((size_t)(NKV * DEPTH_) * DM * 2);
  __bf16* wvt_l = (__bf16*)take((size_t)(NKV * DEPTH_) * DM * 2);
  __bf16* wot_h = (__bf16*)take((size_t)DM * DM * 2);
  __bf16* wot_l = (__bf16*)take((size_t)DM * DM * 2);
  __bf16* vt_h  = (__bf16*)take((size_t)B_ * NKV * DEPTH_ * S_ * 2);
  __bf16* vt_l  = (__bf16*)take((size_t)B_ * NKV * DEPTH_ * S_ * 2);
  float*  attn  = (float*)take((size_t)B_ * S_ * DM * 4);

  const int T = 256;
  // 1. degenerate-RoPE collapse: per-head column sums of Wq / Wk
  colsum_kernel<<<(DM * NH + T - 1) / T, T, 0, stream>>>(Wq, wqsum, DM, NH);
  colsum_kernel<<<(DM * NKV + T - 1) / T, T, 0, stream>>>(Wk, wksum, DM, NKV);
  // 2. Sq / Sk skinny projections
  rowsum_q_kernel<<<(B_ * S_ * NH + T - 1) / T, T, 0, stream>>>(q_in, wqsum, Sq);
  rowsum_k_kernel<<<(B_ * S_ * NKV + T - 1) / T, T, 0, stream>>>(k_in, wksum, Skc);
  // 3. exact softmax max bounds
  minmax_kernel<<<B_ * NKV, T, 0, stream>>>(Skc, skmax, skmin);
  // 4. weight transposes + bf16 hi/lo splits
  convertBt_kernel<<<(DM * NKV * DEPTH_ + T - 1) / T, T, 0, stream>>>(Wv, wvt_h, wvt_l, DM, NKV * DEPTH_);
  convertBt_kernel<<<(DM * DM + T - 1) / T, T, 0, stream>>>(Wo, wot_h, wot_l, DM, DM);
  // 5. V projection straight into the v_cache output region (remapped rows)
  gemm_bf16x3_kernel<<<dim3((NKV * DEPTH_) / 64, (B_ * S_) / 128), T, 0, stream>>>(
      v_in, wvt_h, wvt_l, vcache, B_ * S_, NKV * DEPTH_, DM, 1);
  // 6. k_cache broadcast + zero padding of both caches
  fill_cache_kernel<<<(MAXB * MAXS * NKV * DEPTH_ + T - 1) / T, T, 0, stream>>>(Skc, kcache, vcache);
  // 7. V -> transposed bf16 hi/lo for the attention WMMA B-operand
  convert_vt_kernel<<<(B_ * NKV * DEPTH_ * S_ + T - 1) / T, T, 0, stream>>>(vcache, vt_h, vt_l);
  // 8. fused softmax(attn) @ V
  attention_kernel<<<(B_ * NH * (S_ / 16)) / 8, T, 0, stream>>>(Sq, Skc, skmax, skmin, vt_h, vt_l, attn);
  // 9. output projection
  gemm_bf16x3_kernel<<<dim3(DM / 64, (B_ * S_) / 128), T, 0, stream>>>(
      attn, wot_h, wot_l, out, B_ * S_, DM, DM, 0);
}
